// NearestNeighborEncoder_79869211836496
// MI455X (gfx1250) — compile-verified, matrix operand order checked by reading, not run on hardware
//
#include <hip/hip_runtime.h>

// ---------------------------------------------------------------------------
// NearestNeighborEncoder for MI455X (gfx1250, wave32, WMMA bf16 + TDM)
// GEMMs + attention matmuls via v_wmma_f32_16x16x32_bf16 (fp32 accumulate).
// GEMM tiles staged by the Tensor Data Mover (tensor_load_to_lds, TENSORcnt),
// double-buffered in LDS; norms / softmax / rope in fp32 VALU.
// ---------------------------------------------------------------------------

#define B_  4
#define C_  16
#define N_  2
#define NL_ 128
#define D_  1024
#define H_  16
#define DK_ 64
#define DFF_ 4096
#define CHUNK_ 64
#define ROWS_E (B_*C_*N_*NL_)   // 16384
#define ROWS_H (B_*C_*CHUNK_)   // 4096
#define BF_    (B_*C_*N_)       // 128 sequences for self-attn / cross queries
#define SCALE_ 0.125f           // 1/sqrt(64)

typedef __attribute__((ext_vector_type(16))) __bf16 v16bf;
typedef __attribute__((ext_vector_type(8)))  __bf16 v8bf;
typedef __attribute__((ext_vector_type(8)))  float  v8f;
typedef __attribute__((ext_vector_type(4)))  unsigned int u32x4;
typedef __attribute__((ext_vector_type(8)))  int i32x8;
typedef __attribute__((ext_vector_type(4)))  int i32x4;

#if __has_builtin(__builtin_amdgcn_tensor_load_to_lds) && \
    __has_builtin(__builtin_amdgcn_s_wait_tensorcnt)
#define USE_TDM 1
#else
#define USE_TDM 0
#endif

static __device__ __forceinline__ __bf16 f2bf(float f) {
  unsigned u = __float_as_uint(f);
  unsigned r = (u + 0x7FFFu + ((u >> 16) & 1u)) >> 16;   // round-to-nearest-even
  unsigned short s = (unsigned short)r;
  return __builtin_bit_cast(__bf16, s);
}

// Load a 16x32 bf16 WMMA fragment (A-layout) from row-major [rows][stride]
// storage. Per CDNA5 ISA: lanes 0-15 hold row M=lane with K {0..7,16..23},
// lanes 16-31 hold row M=lane-16 with K {8..15,24..31}. B fragments are the
// mirror (lane -> column N), so the same loader works on B^T-major storage.
static __device__ __forceinline__ v16bf load_frag(const __bf16* base, int stride,
                                                  int row0, int k0) {
  int lane = threadIdx.x & 31;
  int r  = row0 + (lane & 15);
  int kb = k0 + ((lane >> 4) << 3);
  const __bf16* p = base + r * stride + kb;
  v8bf lo = *(const v8bf*)(p);
  v8bf hi = *(const v8bf*)(p + 16);
  return __builtin_shufflevector(lo, hi, 0,1,2,3,4,5,6,7,8,9,10,11,12,13,14,15);
}

#if USE_TDM
// TDM: DMA one 128-row x 32-elem bf16 tile (row pitch = pitch elems) from
// global into LDS at byte offset lds_off, with 4 DWORDs of padding inserted
// after every 16 DWORDs (64B row) -> padded LDS stride of 40 bf16 elements.
// D# layout per CDNA5 ISA ch.8 (group0: count/lds/global/type, group1: dims).
// This toolchain (clang-23 / therock-10.0) exposes the 6-arg builtin:
//   (u32x4 g0, i32x8 g1, i32x4, i32x4, i32x8, i32 cpol)
// where the trailing vector args carry the (unused for 2-D tiles) group2/3/
// extended descriptor words -- zero-filled here.
static __device__ __forceinline__ void tdm_load_tile(const __bf16* gsrc,
                                                     unsigned lds_off,
                                                     int pitch) {
  unsigned long long ga = (unsigned long long)(size_t)gsrc;
  u32x4 g0;
  g0.x = 1u;                                            // count=1 (valid, user)
  g0.y = lds_off;                                       // lds_addr (bytes)
  g0.z = (unsigned)(ga & 0xFFFFFFFFu);                  // global_addr[31:0]
  g0.w = (unsigned)((ga >> 32) & 0x01FFFFFFu)           // global_addr[56:32]
         | (2u << 30);                                  // type = 2 ("image")
  i32x8 g1;
  g1[0] = (int)((1u << 16)      // data_size = 1 -> 2 bytes
              | (1u << 20)      // pad_enable
              | (3u << 22)      // pad_interval: 16 DWORDs
              | (3u << 25));    // pad_amount:   4 DWORDs
  g1[1] = (int)(32u << 16);     // tensor_dim0 = 32
  g1[2] = (int)(128u << 16);    // tensor_dim0 hi | tensor_dim1 = 128
  g1[3] = (int)(32u << 16);     // tensor_dim1 hi | tile_dim0 = 32
  g1[4] = 128;                  // tile_dim1 = 128, tile_dim2 = 0
  g1[5] = pitch;                // tensor_dim0_stride[31:0] (elements)
  g1[6] = 0;                    // stride hi | dim1_stride lo
  g1[7] = 0;
  i32x4 z4 = {0, 0, 0, 0};
  i32x8 z8 = {0, 0, 0, 0, 0, 0, 0, 0};
  __builtin_amdgcn_tensor_load_to_lds(g0, g1, z4, z4, z8, 0);
}
#endif

// ---------------------------------------------------------------------------
// Elementwise / prep kernels
// ---------------------------------------------------------------------------

// W[K][N] fp32  ->  Wt[N][K] bf16
__global__ void __launch_bounds__(256)
k_conv_transpose(const float* __restrict__ w, __bf16* __restrict__ wt,
                 int K, int N) {
  long long idx = (long long)blockIdx.x * 256 + threadIdx.x;
  long long total = (long long)K * N;
  if (idx >= total) return;
  int n = (int)(idx / K);
  int k = (int)(idx % K);
  wt[idx] = f2bf(w[(long long)k * N + n]);
}

__global__ void __launch_bounds__(256)
k_conv_bf16(const float* __restrict__ x, __bf16* __restrict__ y, long long total) {
  long long idx = (long long)blockIdx.x * 256 + threadIdx.x;
  if (idx < total) y[idx] = f2bf(x[idx]);
}

// one block per row; D = 1024
__global__ void __launch_bounds__(256)
k_rmsnorm_bf16(const float* __restrict__ x, const float* __restrict__ g,
               __bf16* __restrict__ out) {
  __shared__ float red[8];
  long long row = blockIdx.x;
  const float* xr = x + row * D_;
  int t = threadIdx.x;
  float s = 0.f;
  float v[4];
  #pragma unroll
  for (int i = 0; i < 4; ++i) { v[i] = xr[t + i * 256]; s += v[i] * v[i]; }
  #pragma unroll
  for (int o = 16; o > 0; o >>= 1) s += __shfl_xor(s, o);
  if ((t & 31) == 0) red[t >> 5] = s;
  __syncthreads();
  float tot = 0.f;
  #pragma unroll
  for (int i = 0; i < 8; ++i) tot += red[i];
  float inv = rsqrtf(tot * (1.0f / D_) + 1e-8f);
  __bf16* orow = out + row * D_;
  #pragma unroll
  for (int i = 0; i < 4; ++i) {
    int c = t + i * 256;
    orow[c] = f2bf(v[i] * inv * g[c]);
  }
}

// RoPE over [rows, H, DK] fp32 -> bf16 (position = row % NL)
__global__ void __launch_bounds__(256)
k_rope_bf16(const float* __restrict__ x, __bf16* __restrict__ out, long long total) {
  long long idx = (long long)blockIdx.x * 256 + threadIdx.x;
  if (idx >= total) return;
  int within = (int)(idx & (D_ - 1));
  long long row = idx >> 10;
  int hd = within & (DK_ - 1);
  int pos = (int)(row & (NL_ - 1));
  int j = hd & 31;
  float theta = __powf(10000.0f, -(float)(2 * j) / (float)DK_);
  float ang = (float)pos * theta;
  float cs = __cosf(ang), sn = __sinf(ang);
  float xv = x[idx];
  float partner = (hd < 32) ? x[idx + 32] : x[idx - 32];
  float nh = (hd < 32) ? -partner : partner;
  out[idx] = f2bf(xv * cs + nh * sn);
}

// ---------------------------------------------------------------------------
// GEMM: C[M][N] = A[M][K](bf16) * Wt[N][K](bf16)^T + bias (+res) (+relu)
// 128x128 block tile, BK=32, 256 threads = 8 waves, wave = 16 rows x 128 cols.
// Tiles staged via TDM (double-buffered) when available.
// ---------------------------------------------------------------------------
#define BKP 40               // 32 + 8 bf16 pad (16B aligned, dodges conflicts)
#define TILE_ELEMS (128 * BKP)
#define BUF_ELEMS  (2 * TILE_ELEMS)          // As + Bs per buffer
#define GEMM_LDS   (2 * BUF_ELEMS * 2)       // double buffered, bytes

__global__ void __launch_bounds__(256)
k_gemm_bf16(const __bf16* __restrict__ A, const __bf16* __restrict__ Wt,
            const float* __restrict__ bias, const float* __restrict__ res,
            float* __restrict__ outF, __bf16* __restrict__ outB,
            int M, int N, int K, int relu) {
  extern __shared__ char smem[];
  __bf16* lds = (__bf16*)smem;

  const int tid  = threadIdx.x;
  const int wave = tid >> 5;
  const int lane = tid & 31;
  const int half = lane >> 4;
  const int m0g  = blockIdx.y * 128;
  const int n0g  = blockIdx.x * 128;

  v8f acc[8];
  #pragma unroll
  for (int i = 0; i < 8; ++i) acc[i] = (v8f)(0.0f);

  const int ksteps = K >> 5;

#if USE_TDM
  const unsigned lds_base = (unsigned)(size_t)(void*)lds;
  // preload tile 0 into buffer 0 (wave 0 drives the TDM)
  if (wave == 0) {
    tdm_load_tile(A  + (long long)m0g * K, lds_base, K);
    tdm_load_tile(Wt + (long long)n0g * K, lds_base + TILE_ELEMS * 2, K);
    __builtin_amdgcn_s_wait_tensorcnt(0);
  }
  __syncthreads();

  for (int kt = 0; kt < ksteps; ++kt) {
    const int cur = kt & 1;
    if (wave == 0 && kt + 1 < ksteps) {     // issue next tile into other buffer
      const int kk = (kt + 1) << 5;
      unsigned dst = lds_base + (cur ^ 1) * (BUF_ELEMS * 2);
      tdm_load_tile(A  + (long long)m0g * K + kk, dst, K);
      tdm_load_tile(Wt + (long long)n0g * K + kk, dst + TILE_ELEMS * 2, K);
    }
    const __bf16* As = lds + cur * BUF_ELEMS;
    const __bf16* Bs = As + TILE_ELEMS;

    v16bf a = load_frag(As, BKP, wave << 4, 0);
    v16bf bfr[8];
    #pragma unroll
    for (int nt = 0; nt < 8; ++nt) bfr[nt] = load_frag(Bs, BKP, nt << 4, 0);
    #pragma unroll
    for (int nt = 0; nt < 8; ++nt)
      acc[nt] = __builtin_amdgcn_wmma_f32_16x16x32_bf16(
          false, a, false, bfr[nt], (short)0, acc[nt], false, false);

    if (wave == 0) __builtin_amdgcn_s_wait_tensorcnt(0);
    __syncthreads();
  }
#else
  __bf16* As = lds;
  __bf16* Bs = lds + TILE_ELEMS;
  for (int kt = 0; kt < ksteps; ++kt) {
    const int kk = kt << 5;
    #pragma unroll
    for (int rep = 0; rep < 2; ++rep) {
      int c   = tid + rep * 256;
      int row = c >> 2;
      int kc  = (c & 3) * 8;
      *(v8bf*)(As + row * BKP + kc) =
          *(const v8bf*)(A + (long long)(m0g + row) * K + kk + kc);
      *(v8bf*)(Bs + row * BKP + kc) =
          *(const v8bf*)(Wt + (long long)(n0g + row) * K + kk + kc);
    }
    if (kt + 1 < ksteps)
      __builtin_prefetch(A + (long long)(m0g + (tid >> 1)) * K + kk + 32, 0, 1);
    __syncthreads();

    v16bf a = load_frag(As, BKP, wave << 4, 0);
    v16bf bfr[8];
    #pragma unroll
    for (int nt = 0; nt < 8; ++nt) bfr[nt] = load_frag(Bs, BKP, nt << 4, 0);
    #pragma unroll
    for (int nt = 0; nt < 8; ++nt)
      acc[nt] = __builtin_amdgcn_wmma_f32_16x16x32_bf16(
          false, a, false, bfr[nt], (short)0, acc[nt], false, false);
    __syncthreads();
  }
#endif

  // epilogue: bias, relu, residual, fp32/bf16 stores
  #pragma unroll
  for (int nt = 0; nt < 8; ++nt) {
    int n = n0g + (nt << 4) + (lane & 15);
    float bv = bias ? bias[n] : 0.0f;
    #pragma unroll
    for (int r = 0; r < 8; ++r) {
      int m = m0g + (wave << 4) + r + (half << 3);
      float v = acc[nt][r] + bv;
      if (relu) v = fmaxf(v, 0.0f);
      long long o = (long long)m * N + n;
      if (res)  v += res[o];
      if (outF) outF[o] = v;
      if (outB) outB[o] = f2bf(v);
    }
  }
}

// ---------------------------------------------------------------------------
// Fused attention: one block per (batch, head). S = Q K^T (WMMA), fp32
// softmax in LDS, O = P V (WMMA), bf16 output. Lq == 128 always.
// ---------------------------------------------------------------------------
#define QS_STR 72
#define VT_STR 136
#define S_STR  132
#define P_STR  136
#define ATTN_LDS (128*QS_STR*2 /*Q*/ + 128*QS_STR*2 /*K*/ + 64*VT_STR*2 /*Vt*/ \
                  + 128*S_STR*4 /*S*/ + 128*P_STR*2 /*P*/)

template <int LK>
__global__ void __launch_bounds__(256)
k_attn_wmma(const __bf16* __restrict__ q, const __bf16* __restrict__ k,
            const __bf16* __restrict__ v, __bf16* __restrict__ o, int kvdiv) {
  extern __shared__ char smem[];
  __bf16* Qs = (__bf16*)smem;                      // [128][QS_STR]
  __bf16* Ks = Qs + 128 * QS_STR;                  // [LK][QS_STR]
  __bf16* Vt = Ks + 128 * QS_STR;                  // [64][VT_STR]  (V^T)
  float*  S  = (float*)(Vt + 64 * VT_STR);         // [128][S_STR]
  __bf16* P  = (__bf16*)(S + 128 * S_STR);         // [128][P_STR]

  const int tid  = threadIdx.x;
  const int wave = tid >> 5;
  const int lane = tid & 31;
  const int half = lane >> 4;
  const int qb   = blockIdx.x;
  const int head = blockIdx.y;
  const int kvb  = qb / kvdiv;

  // stage Q (128x64), K (LKx64), V^T (64xLK)
  for (int c = tid; c < 128 * 8; c += 256) {
    int row = c >> 3, kc = (c & 7) * 8;
    *(v8bf*)(Qs + row * QS_STR + kc) =
        *(const v8bf*)(q + ((long long)(qb * 128 + row) * D_) + head * DK_ + kc);
  }
  for (int c = tid; c < LK * 8; c += 256) {
    int row = c >> 3, kc = (c & 7) * 8;
    *(v8bf*)(Ks + row * QS_STR + kc) =
        *(const v8bf*)(k + ((long long)(kvb * LK + row) * D_) + head * DK_ + kc);
  }
  for (int i = tid; i < LK * 64; i += 256) {
    int l = i >> 6, dk = i & 63;
    Vt[dk * VT_STR + l] = v[((long long)(kvb * LK + l) * D_) + head * DK_ + dk];
  }
  __syncthreads();

  // --- S = Q K^T * scale ---
  {
    v8f acc[LK / 16];
    #pragma unroll
    for (int i = 0; i < LK / 16; ++i) acc[i] = (v8f)(0.0f);
    #pragma unroll
    for (int kt = 0; kt < 2; ++kt) {                 // DK = 64 -> 2 x K32
      v16bf a = load_frag(Qs, QS_STR, wave << 4, kt << 5);
      v16bf bfr[LK / 16];
      #pragma unroll
      for (int nt = 0; nt < LK / 16; ++nt)
        bfr[nt] = load_frag(Ks, QS_STR, nt << 4, kt << 5);
      #pragma unroll
      for (int nt = 0; nt < LK / 16; ++nt)
        acc[nt] = __builtin_amdgcn_wmma_f32_16x16x32_bf16(
            false, a, false, bfr[nt], (short)0, acc[nt], false, false);
    }
    #pragma unroll
    for (int nt = 0; nt < LK / 16; ++nt) {
      int n = (nt << 4) + (lane & 15);
      #pragma unroll
      for (int r = 0; r < 8; ++r) {
        int m = (wave << 4) + r + (half << 3);
        S[m * S_STR + n] = acc[nt][r] * SCALE_;
      }
    }
  }
  __syncthreads();

  // --- softmax rows (threads 0..127) ---
  if (tid < 128) {
    float* srow = S + tid * S_STR;
    float mx = -3.0e38f;
    for (int j = 0; j < LK; ++j) mx = fmaxf(mx, srow[j]);
    float sum = 0.f;
    for (int j = 0; j < LK; ++j) { float e = __expf(srow[j] - mx); srow[j] = e; sum += e; }
    float inv = 1.0f / sum;
    __bf16* prow = P + tid * P_STR;
    for (int j = 0; j < LK; ++j) prow[j] = f2bf(srow[j] * inv);
  }
  __syncthreads();

  // --- O = P V ---
  {
    v8f acc[4];
    #pragma unroll
    for (int i = 0; i < 4; ++i) acc[i] = (v8f)(0.0f);
    #pragma unroll
    for (int kt = 0; kt < LK / 32; ++kt) {
      v16bf a = load_frag(P, P_STR, wave << 4, kt << 5);
      v16bf bfr[4];
      #pragma unroll
      for (int nt = 0; nt < 4; ++nt)
        bfr[nt] = load_frag(Vt, VT_STR, nt << 4, kt << 5);
      #pragma unroll
      for (int nt = 0; nt < 4; ++nt)
        acc[nt] = __builtin_amdgcn_wmma_f32_16x16x32_bf16(
            false, a, false, bfr[nt], (short)0, acc[nt], false, false);
    }
    #pragma unroll
    for (int nt = 0; nt < 4; ++nt) {
      int n = (nt << 4) + (lane & 15);
      #pragma unroll
      for (int r = 0; r < 8; ++r) {
        int m = (wave << 4) + r + (half << 3);
        o[((long long)(qb * 128 + m) * D_) + head * DK_ + n] = f2bf(acc[nt][r]);
      }
    }
  }
}

// ---------------------------------------------------------------------------
// Host orchestration
// ---------------------------------------------------------------------------
static inline int ceil_div_ll(long long a, int b) { return (int)((a + b - 1) / b); }

extern "C" void kernel_launch(void* const* d_in, const int* in_sizes, int n_in,
                              void* d_out, int out_size, void* d_ws, size_t ws_size,
                              hipStream_t stream) {
  (void)in_sizes; (void)n_in; (void)out_size; (void)ws_size;

  // --- input indices (jax tree-flatten: sorted dict keys) ---
  // 0:e 1:h ; attn[p] @ 2+9p: k.b,k.w,norm,o.b,o.w,q.b,q.w,v.b,v.w
  // ca @ 29 (same) ; ffw[p] @ 38+5p: l1.b,l1.w,l2.b,l2.w,norm ; 53: norm_h
  const float* eIn = (const float*)d_in[0];
  const float* hIn = (const float*)d_in[1];
  auto F = [&](int i) { return (const float*)d_in[i]; };
  const int ATTN0 = 2, CA0 = 29, FFW0 = 38, NORMH = 53;

  // --- workspace carve ---
  size_t off = 0;
  auto alloc = [&](size_t bytes) -> void* {
    size_t cur = (off + 255) & ~(size_t)255;
    off = cur + bytes;
    return (void*)((char*)d_ws + cur);
  };
  __bf16* wt_attn[3][4];   // q,k,v,o  (each [1024][1024] bf16, N-major)
  for (int p = 0; p < 3; ++p)
    for (int j = 0; j < 4; ++j)
      wt_attn[p][j] = (__bf16*)alloc((size_t)D_ * D_ * 2);
  __bf16* wt_ca[4];
  for (int j = 0; j < 4; ++j) wt_ca[j] = (__bf16*)alloc((size_t)D_ * D_ * 2);
  __bf16* wt_ffw1[3]; __bf16* wt_ffw2[3];
  for (int p = 0; p < 3; ++p) {
    wt_ffw1[p] = (__bf16*)alloc((size_t)DFF_ * D_ * 2);   // [4096][1024]
    wt_ffw2[p] = (__bf16*)alloc((size_t)D_ * DFF_ * 2);   // [1024][4096]
  }
  __bf16* xn   = (__bf16*)alloc((size_t)ROWS_E * D_ * 2);
  __bf16* qbuf = (__bf16*)alloc((size_t)ROWS_E * D_ * 2);
  __bf16* kbuf = (__bf16*)alloc((size_t)ROWS_E * D_ * 2);
  __bf16* vbuf = (__bf16*)alloc((size_t)ROWS_E * D_ * 2);
  __bf16* obuf = (__bf16*)alloc((size_t)ROWS_E * D_ * 2);
  __bf16* mid  = (__bf16*)alloc((size_t)ROWS_E * DFF_ * 2);
  __bf16* hnorm= (__bf16*)alloc((size_t)ROWS_H * D_ * 2);
  __bf16* hk   = (__bf16*)alloc((size_t)ROWS_H * D_ * 2);
  __bf16* hv   = (__bf16*)alloc((size_t)ROWS_H * D_ * 2);
  float*  scr  = (float*) alloc((size_t)ROWS_E * D_ * 4);
  float*  ecur = (float*) alloc((size_t)ROWS_E * D_ * 4);

  (void)hipFuncSetAttribute((const void*)k_attn_wmma<128>,
                            hipFuncAttributeMaxDynamicSharedMemorySize, ATTN_LDS);
  (void)hipFuncSetAttribute((const void*)k_attn_wmma<64>,
                            hipFuncAttributeMaxDynamicSharedMemorySize, ATTN_LDS);

  auto gemm = [&](const __bf16* A, const __bf16* Wt, const float* bias,
                  const float* res, float* outF, __bf16* outB,
                  int M, int N, int K, int relu) {
    dim3 grid(N / 128, M / 128);
    k_gemm_bf16<<<grid, 256, GEMM_LDS, stream>>>(A, Wt, bias, res, outF, outB,
                                                 M, N, K, relu);
  };
  auto transpose = [&](const float* w, __bf16* wt, int K, int N) {
    long long tot = (long long)K * N;
    k_conv_transpose<<<ceil_div_ll(tot, 256), 256, 0, stream>>>(w, wt, K, N);
  };
  auto conv = [&](const float* x, __bf16* y, long long tot) {
    k_conv_bf16<<<ceil_div_ll(tot, 256), 256, 0, stream>>>(x, y, tot);
  };
  auto rope = [&](const float* x, __bf16* y) {
    long long tot = (long long)ROWS_E * D_;
    k_rope_bf16<<<ceil_div_ll(tot, 256), 256, 0, stream>>>(x, y, tot);
  };

  // --- weight conversion (fp32 [K][N] -> bf16 W^T [N][K]) ---
  for (int p = 0; p < 3; ++p) {
    int b = ATTN0 + 9 * p;
    transpose(F(b + 6), wt_attn[p][0], D_, D_);  // q.w
    transpose(F(b + 1), wt_attn[p][1], D_, D_);  // k.w
    transpose(F(b + 8), wt_attn[p][2], D_, D_);  // v.w
    transpose(F(b + 4), wt_attn[p][3], D_, D_);  // o.w
  }
  transpose(F(CA0 + 6), wt_ca[0], D_, D_);
  transpose(F(CA0 + 1), wt_ca[1], D_, D_);
  transpose(F(CA0 + 8), wt_ca[2], D_, D_);
  transpose(F(CA0 + 4), wt_ca[3], D_, D_);
  for (int p = 0; p < 3; ++p) {
    int b = FFW0 + 5 * p;
    transpose(F(b + 1), wt_ffw1[p], D_, DFF_);   // lin1.w [1024][4096] -> [4096][1024]
    transpose(F(b + 3), wt_ffw2[p], DFF_, D_);   // lin2.w [4096][1024] -> [1024][4096]
  }

  // h_split rms-norm -> bf16 (ROWS_H rows)
  k_rmsnorm_bf16<<<ROWS_H, 256, 0, stream>>>(hIn, F(NORMH), hnorm);

  // working copy of e
  (void)hipMemcpyAsync(ecur, eIn, (size_t)ROWS_E * D_ * 4,
                       hipMemcpyDeviceToDevice, stream);

  for (int p = 0; p < 3; ++p) {
    int b = ATTN0 + 9 * p;
    // ---- self attention ----
    k_rmsnorm_bf16<<<ROWS_E, 256, 0, stream>>>(ecur, F(b + 2), xn);
    gemm(xn, wt_attn[p][0], F(b + 5), nullptr, scr, nullptr, ROWS_E, D_, D_, 0);
    rope(scr, qbuf);
    gemm(xn, wt_attn[p][1], F(b + 0), nullptr, scr, nullptr, ROWS_E, D_, D_, 0);
    rope(scr, kbuf);
    gemm(xn, wt_attn[p][2], F(b + 7), nullptr, scr, nullptr, ROWS_E, D_, D_, 0);
    conv(scr, vbuf, (long long)ROWS_E * D_);
    k_attn_wmma<128><<<dim3(BF_, H_), 256, ATTN_LDS, stream>>>(qbuf, kbuf, vbuf,
                                                               obuf, 1);
    gemm(obuf, wt_attn[p][3], F(b + 3), ecur, ecur, nullptr, ROWS_E, D_, D_, 0);

    // ---- cross attention (layer 1) ----
    if (p == 1) {
      int cb = CA0;
      k_rmsnorm_bf16<<<ROWS_E, 256, 0, stream>>>(ecur, F(cb + 2), xn);
      gemm(xn, wt_ca[0], F(cb + 5), nullptr, scr, nullptr, ROWS_E, D_, D_, 0);
      conv(scr, qbuf, (long long)ROWS_E * D_);                 // no rope
      gemm(hnorm, wt_ca[1], F(cb + 0), nullptr, scr, nullptr, ROWS_H, D_, D_, 0);
      conv(scr, hk, (long long)ROWS_H * D_);
      gemm(hnorm, wt_ca[2], F(cb + 7), nullptr, scr, nullptr, ROWS_H, D_, D_, 0);
      conv(scr, hv, (long long)ROWS_H * D_);
      k_attn_wmma<64><<<dim3(BF_, H_), 256, ATTN_LDS, stream>>>(qbuf, hk, hv,
                                                                obuf, N_);
      gemm(obuf, wt_ca[3], F(cb + 3), ecur, ecur, nullptr, ROWS_E, D_, D_, 0);
    }

    // ---- FFW ----
    int fb = FFW0 + 5 * p;
    k_rmsnorm_bf16<<<ROWS_E, 256, 0, stream>>>(ecur, F(fb + 4), xn);
    gemm(xn,  wt_ffw1[p], F(fb + 0), nullptr, nullptr, mid, ROWS_E, DFF_, D_, 1);
    gemm(mid, wt_ffw2[p], F(fb + 2), ecur, ecur, nullptr, ROWS_E, D_, DFF_, 0);
  }

  (void)hipMemcpyAsync(d_out, ecur, (size_t)ROWS_E * D_ * 4,
                       hipMemcpyDeviceToDevice, stream);
}
